// DecoderBlock_49168785604684
// MI455X (gfx1250) — matrix operand order repaired and checked by reading, not compiled, not run on hardware
//
#include <hip/hip_runtime.h>

typedef __attribute__((ext_vector_type(16))) _Float16 v16h;
typedef __attribute__((ext_vector_type(8)))  _Float16 v8h;
typedef __attribute__((ext_vector_type(8)))  float    v8f;
typedef __attribute__((ext_vector_type(4)))  int      v4i;

#define BM 128
#define BN 256
#define BK 32
#define LDA 40   // LDS pitch in halves: 80 bytes -> 16B aligned, bank-friendly

#define GLOBAL_AS __attribute__((address_space(1)))
#define LDS_AS    __attribute__((address_space(3)))

#if __has_builtin(__builtin_amdgcn_global_load_async_to_lds_b128)
#define HAVE_ASYNC 1
#else
#define HAVE_ASYNC 0
#endif

__device__ __forceinline__ void async_cp16(void* lds, const void* g) {
#if HAVE_ASYNC
    __builtin_amdgcn_global_load_async_to_lds_b128(
        (GLOBAL_AS v4i*)g, (LDS_AS v4i*)lds, 0, 0);
#else
    (void)lds; (void)g;
#endif
}
__device__ __forceinline__ void async_wait() {
#if HAVE_ASYNC
#if __has_builtin(__builtin_amdgcn_s_wait_asynccnt)
    __builtin_amdgcn_s_wait_asynccnt(0);
#else
    asm volatile("s_wait_asynccnt 0x0" ::: "memory");
#endif
#endif
}

template<typename T> struct __align__(16) Chunk8 { T v[8]; };

// ---- WMMA fragment loaders (wave32, CDNA5 ISA 7.12.2 layouts) --------------
// 16-bit A 16x32: lanes 0-15 row=lane, halves0-7=K0-7, halves8-15=K16-23;
// lanes 16-31 row=lane-16, halves0-7=K8-15, halves8-15=K24-31.
__device__ __forceinline__ v16h frag_a(const _Float16* s, int row0, int lane) {
    int r  = row0 + (lane & 15);
    int kb = (lane >> 4) << 3;
    const _Float16* p = s + r * LDA + kb;
    v8h lo = *(const v8h*)p;
    v8h hi = *(const v8h*)(p + 16);
    return __builtin_shufflevector(lo, hi, 0,1,2,3,4,5,6,7,8,9,10,11,12,13,14,15);
}
// 16-bit B 32x16: lanes 0-15 col=lane, halves0-7=K0-7, halves8-15=K16-23; lanes 16-31 col=lane-16, halves0-7=K8-15, halves8-15=K24-31.
__device__ __forceinline__ v16h frag_b(const _Float16* s, int col0, int lane) {
    int c  = col0 + (lane & 15);
    int kb = (lane >> 4) << 3;
    const _Float16* p = s + c * LDA + kb;
    v8h lo = *(const v8h*)p;
    v8h hi = *(const v8h*)(p + 16);
    return __builtin_shufflevector(lo, hi, 0,1,2,3,4,5,6,7,8,9,10,11,12,13,14,15);
}

// ---- Generic tiled WMMA GEMM, double-buffered LDS --------------------------
// C[m,n] = act( scale * sum_k A[m,k]*Bsel(k,n) + bias[n] )
//   BT=true : B is [N,K] row-major (A @ B^T); BT=false: B is [K,N] (A @ B)
//   MASK: causal, col > row -> -inf (after scale, no bias)
template<typename TA, typename TB, bool BT, typename TC,
         bool BIAS, bool GELU, bool MASK>
__global__ __launch_bounds__(256)
void wmma_gemm(const TA* __restrict__ A, const TB* __restrict__ B,
               const float* __restrict__ bias, TC* __restrict__ C,
               int Mdim, int Ndim, int Kdim,
               long long sA, long long sB, long long sC, float scale) {
    const int m0 = blockIdx.y * BM;
    const int n0 = blockIdx.x * BN;
    const TA* Ab = A + (long long)blockIdx.z * sA;
    const TB* Bb = B + (long long)blockIdx.z * sB;
    TC*       Cb = C + (long long)blockIdx.z * sC;

    // tiles fully above the causal diagonal: fill -inf, done
    if (MASK && n0 > m0 + (BM - 1)) {
        for (int idx = threadIdx.x; idx < BM * BN; idx += 256) {
            int r = idx >> 8, c = idx & (BN - 1);
            Cb[(long long)(m0 + r) * Ndim + n0 + c] = (TC)(-INFINITY);
        }
        return;
    }

    constexpr bool A16 = (sizeof(TA) == 2);
    constexpr bool B16 = (sizeof(TB) == 2);
    constexpr bool ASY_A = HAVE_ASYNC && A16;          // async byte-copy path
    constexpr bool ASY_B = HAVE_ASYNC && B16 && BT;
    constexpr bool ANY_ASY = ASY_A || ASY_B;

    __shared__ __align__(16) _Float16 As[2][BM * LDA];
    __shared__ __align__(16) _Float16 Bs[2][BN * LDA];

    const int tid  = threadIdx.x;
    const int lane = tid & 31;
    const int wid  = tid >> 5;
    const int wm   = (wid >> 2) * 64;   // 2 waves in M
    const int wn   = (wid & 3) * 64;    // 4 waves in N

    // per-thread staging coordinates
    int arow[2], acg[2], brow[4], bcg[4], bk_[4], bng[4];
#pragma unroll
    for (int c = 0; c < 2; ++c) { int l = tid + 256 * c; arow[c] = l >> 2; acg[c] = (l & 3) << 3; }
#pragma unroll
    for (int c = 0; c < 4; ++c) {
        int l = tid + 256 * c;
        brow[c] = l >> 2;  bcg[c] = (l & 3) << 3;     // BT: [n][k-group]
        bk_[c]  = l >> 5;  bng[c] = (l & 31) << 3;    // !BT: [k][n-group]
    }

    v8f acc[4][4];
#pragma unroll
    for (int i = 0; i < 4; ++i)
#pragma unroll
        for (int j = 0; j < 4; ++j) acc[i][j] = (v8f)(0.0f);

    Chunk8<TA> apre[2];
    Chunk8<TB> bpre[4];

    const int nk = Kdim >> 5;

    // ---- helpers -----------------------------------------------------------
    auto issueA = [&](int buf, int k0) {
#pragma unroll
        for (int c = 0; c < 2; ++c) {
            const TA* src = Ab + (long long)(m0 + arow[c]) * Kdim + k0 + acg[c];
            if (ASY_A) async_cp16(&As[buf][arow[c] * LDA + acg[c]], src);
            else       apre[c] = *(const Chunk8<TA>*)src;
        }
    };
    auto commitA = [&](int buf) {
        if (ASY_A) return;
#pragma unroll
        for (int c = 0; c < 2; ++c) {
            v8h v;
#pragma unroll
            for (int j = 0; j < 8; ++j) v[j] = (_Float16)apre[c].v[j];
            *(v8h*)&As[buf][arow[c] * LDA + acg[c]] = v;
        }
    };
    auto issueB = [&](int buf, int k0) {
#pragma unroll
        for (int c = 0; c < 4; ++c) {
            if (BT) {
                const TB* src = Bb + (long long)(n0 + brow[c]) * Kdim + k0 + bcg[c];
                if (ASY_B) async_cp16(&Bs[buf][brow[c] * LDA + bcg[c]], src);
                else       bpre[c] = *(const Chunk8<TB>*)src;
            } else {
                bpre[c] = *(const Chunk8<TB>*)(Bb + (long long)(k0 + bk_[c]) * Ndim + n0 + bng[c]);
            }
        }
    };
    auto commitB = [&](int buf) {
        if (ASY_B) return;
#pragma unroll
        for (int c = 0; c < 4; ++c) {
            if (BT) {
                v8h v;
#pragma unroll
                for (int j = 0; j < 8; ++j) v[j] = (_Float16)bpre[c].v[j];
                *(v8h*)&Bs[buf][brow[c] * LDA + bcg[c]] = v;
            } else {
#pragma unroll
                for (int j = 0; j < 8; ++j)
                    Bs[buf][(bng[c] + j) * LDA + bk_[c]] = (_Float16)bpre[c].v[j];
            }
        }
    };
    auto compute = [&](int buf) {
        v16h af[4];
#pragma unroll
        for (int i = 0; i < 4; ++i) af[i] = frag_a(&As[buf][0], wm + i * 16, lane);
#pragma unroll
        for (int j = 0; j < 4; ++j) {
            v16h bfr = frag_b(&Bs[buf][0], wn + j * 16, lane);
#pragma unroll
            for (int i = 0; i < 4; ++i)
                acc[i][j] = __builtin_amdgcn_wmma_f32_16x16x32_f16(
                    false, af[i], false, bfr, (short)0, acc[i][j], false, false);
        }
    };

    // ---- prologue: stage tile 0 into buffer 0 ------------------------------
    issueA(0, 0); issueB(0, 0);
    commitA(0);   commitB(0);
    if (ANY_ASY) async_wait();
    __syncthreads();

    // ---- main loop: overlap next tile's loads with current tile's WMMAs ----
    for (int kt = 0; kt < nk; ++kt) {
        const int cur = kt & 1;
        const bool more = (kt + 1) < nk;
        if (more) { issueA(cur ^ 1, (kt + 1) * BK); issueB(cur ^ 1, (kt + 1) * BK); }
        compute(cur);
        if (more) {
            commitA(cur ^ 1); commitB(cur ^ 1);
            if (ANY_ASY) async_wait();
        }
        __syncthreads();
    }

    // ---- epilogue: C/D layout: lane<16 -> M=r, lane>=16 -> M=8+r; N=lane&15
    const bool needmask = MASK && (n0 + (BN - 1) > m0);
    const int rowb = m0 + wm + ((lane >> 4) << 3);
    const int colb = n0 + wn + (lane & 15);
#pragma unroll
    for (int i = 0; i < 4; ++i) {
#pragma unroll
        for (int j = 0; j < 4; ++j) {
            int col = colb + j * 16;
            float bv = BIAS ? bias[col] : 0.0f;
#pragma unroll
            for (int r = 0; r < 8; ++r) {
                int row = rowb + i * 16 + r;
                float v = acc[i][j][r] * scale;
                if (BIAS) v += bv;
                if (MASK && needmask && col > row) v = -INFINITY;
                if (GELU) v = 0.5f * v * (1.0f + erff(v * 0.70710678118f));
                Cb[(long long)row * Ndim + col] = (TC)v;
            }
        }
    }
}

// ---- row softmax over N f32 scores -> f16 probabilities --------------------
__global__ __launch_bounds__(256)
void softmax_rows(const float* __restrict__ S, _Float16* __restrict__ P, int N) {
    const long long row = blockIdx.x;
    const float* srow = S + row * N;
    __shared__ float red[256];
    const int tid = threadIdx.x;

    float mx = -INFINITY;
    for (int c = tid; c < N; c += 256) mx = fmaxf(mx, srow[c]);
    red[tid] = mx; __syncthreads();
    for (int s = 128; s > 0; s >>= 1) {
        if (tid < s) red[tid] = fmaxf(red[tid], red[tid + s]);
        __syncthreads();
    }
    mx = red[0]; __syncthreads();

    float sum = 0.0f;
    for (int c = tid; c < N; c += 256) sum += __expf(srow[c] - mx);
    red[tid] = sum; __syncthreads();
    for (int s = 128; s > 0; s >>= 1) {
        if (tid < s) red[tid] += red[tid + s];
        __syncthreads();
    }
    const float inv = 1.0f / red[0];

    for (int c = tid; c < N; c += 256)
        P[row * N + c] = (_Float16)(__expf(srow[c] - mx) * inv);
}

extern "C" void kernel_launch(void* const* d_in, const int* in_sizes, int n_in,
                              void* d_out, int out_size, void* d_ws, size_t ws_size,
                              hipStream_t stream) {
    (void)in_sizes; (void)n_in; (void)out_size; (void)ws_size;
    constexpr int D = 1024, M = 4096, S = 2048, B = 4;
    constexpr long long ROWS = (long long)B * S;   // 8192

    const float* x  = (const float*)d_in[0];
    const float* Wq = (const float*)d_in[1];
    const float* bq = (const float*)d_in[2];
    const float* Wk = (const float*)d_in[3];
    const float* bk = (const float*)d_in[4];
    const float* Wv = (const float*)d_in[5];
    const float* bv = (const float*)d_in[6];
    const float* Wo = (const float*)d_in[7];
    const float* bo = (const float*)d_in[8];
    const float* Wf = (const float*)d_in[9];
    const float* bf = (const float*)d_in[10];
    float* out = (float*)d_out;

    // workspace carve-up: 16+16+16+64+32 = 144 MB
    _Float16* Qb = (_Float16*)d_ws;                 // [8192,1024] f16
    _Float16* Kb = Qb + ROWS * D;
    _Float16* Vb = Kb + ROWS * D;
    float*    Sc = (float*)(Vb + ROWS * D);         // [4,2048,2048] f32
    _Float16* Pb = (_Float16*)(Sc + (long long)B * S * S);
    _Float16* At = Qb;                              // attn out reuses Q buffer
    _Float16* O1 = Kb;                              // post-Wo reuses K buffer

    dim3 blk(256);
    const float rsq = 0.03125f;                     // 1/sqrt(1024)

    // Q/K/V projections: x @ W^T + b -> f16
    wmma_gemm<float, float, true, _Float16, true, false, false>
        <<<dim3(D / BN, ROWS / BM, 1), blk, 0, stream>>>(
            x, Wq, bq, Qb, (int)ROWS, D, D, 0, 0, 0, 1.0f);
    wmma_gemm<float, float, true, _Float16, true, false, false>
        <<<dim3(D / BN, ROWS / BM, 1), blk, 0, stream>>>(
            x, Wk, bk, Kb, (int)ROWS, D, D, 0, 0, 0, 1.0f);
    wmma_gemm<float, float, true, _Float16, true, false, false>
        <<<dim3(D / BN, ROWS / BM, 1), blk, 0, stream>>>(
            x, Wv, bv, Vb, (int)ROWS, D, D, 0, 0, 0, 1.0f);

    // scores = (Q @ K^T)/32 with causal mask, batched over B (async f16 path)
    wmma_gemm<_Float16, _Float16, true, float, false, false, true>
        <<<dim3(S / BN, S / BM, B), blk, 0, stream>>>(
            Qb, Kb, nullptr, Sc, S, S, D,
            (long long)S * D, (long long)S * D, (long long)S * S, rsq);

    // softmax rows -> P (f16)
    softmax_rows<<<dim3((unsigned)ROWS), blk, 0, stream>>>(Sc, Pb, S);

    // attn = P @ V (B non-transposed)
    wmma_gemm<_Float16, _Float16, false, _Float16, false, false, false>
        <<<dim3(D / BN, S / BM, B), blk, 0, stream>>>(
            Pb, Vb, nullptr, At, S, D, S,
            (long long)S * S, (long long)S * D, (long long)S * D, 1.0f);

    // out1 = attn @ Wo^T + bo
    wmma_gemm<_Float16, float, true, _Float16, true, false, false>
        <<<dim3(D / BN, ROWS / BM, 1), blk, 0, stream>>>(
            At, Wo, bo, O1, (int)ROWS, D, D, 0, 0, 0, 1.0f);

    // ff = gelu(out1 @ Wf^T + bf) -> d_out (f32)
    wmma_gemm<_Float16, float, true, float, true, true, false>
        <<<dim3(M / BN, ROWS / BM, 1), blk, 0, stream>>>(
            O1, Wf, bf, out, (int)ROWS, M, D, 0, 0, 0, 1.0f);
}
